// PointNet2Encoder_70557722739270
// MI455X (gfx1250) — matrix operand order repaired and checked by reading, not compiled, not run on hardware
//
#include <hip/hip_runtime.h>

typedef __attribute__((ext_vector_type(16))) _Float16 v16h;
typedef __attribute__((ext_vector_type(8)))  _Float16 v8h;
typedef __attribute__((ext_vector_type(8)))  float    v8f;

#define CDIV(a,b) (((a)+(b)-1)/(b))

// ---------------------------------------------------------------------------
// Utility kernels
// ---------------------------------------------------------------------------
__global__ void fill_f32(float* __restrict__ p, float v, int n) {
    int i = blockIdx.x * blockDim.x + threadIdx.x;
    if (i < n) p[i] = v;
}

// Convert weight (Cout,Cin) f32 -> (Cout,Kpad) f16, zero-padded K
__global__ void conv_w_f16(const float* __restrict__ W, _Float16* __restrict__ o,
                           int cout, int cin, int kpad) {
    int i = blockIdx.x * blockDim.x + threadIdx.x;
    int total = cout * kpad;
    if (i >= total) return;
    int r = i / kpad, k = i % kpad;
    o[i] = (k < cin) ? (_Float16)W[(size_t)r * cin + k] : (_Float16)0.f;
}

// (B,3,N) -> (B,N,3)
__global__ void transpose_xyz(const float* __restrict__ in, float* __restrict__ out,
                              int B, int N) {
    int i = blockIdx.x * blockDim.x + threadIdx.x;
    int total = B * N * 3;
    if (i >= total) return;
    int b = i / (N * 3);
    int r = i % (N * 3);
    int n = r / 3, c = r % 3;
    out[i] = in[(size_t)b * 3 * N + (size_t)c * N + n];
}

// ---------------------------------------------------------------------------
// Farthest point sampling. One block (256 threads) per batch; N = PPT*256.
// Matches jax ref: idx[0]=0, then argmax of running min-distance (first-index ties).
// ---------------------------------------------------------------------------
template <int PPT>
__global__ void fps_kernel(const float* __restrict__ x, int N, int S, int* __restrict__ out) {
    const int T = 256;
    int b = blockIdx.x, t = threadIdx.x;
    const float* xb = x + (size_t)b * N * 3;
    int* ob = out + (size_t)b * S;
    float dist[PPT];
#pragma unroll
    for (int i = 0; i < PPT; i++) dist[i] = 1e10f;
    __shared__ float sv[T];
    __shared__ int   si[T];
    __shared__ int   sfar;
    int far = 0;
    if (t == 0) ob[0] = 0;
    for (int it = 1; it < S; ++it) {
        float cx = xb[far * 3 + 0], cy = xb[far * 3 + 1], cz = xb[far * 3 + 2];
        float best = -1.f;
        int bi = 0x7fffffff;
#pragma unroll
        for (int i = 0; i < PPT; i++) {
            int p = t + i * T;
            float dx = xb[p * 3 + 0] - cx;
            float dy = xb[p * 3 + 1] - cy;
            float dz = xb[p * 3 + 2] - cz;
            float d = dx * dx + dy * dy + dz * dz;
            d = fminf(dist[i], d);
            dist[i] = d;
            if (d > best) { best = d; bi = p; }
        }
        sv[t] = best; si[t] = bi;
        __syncthreads();
        for (int stp = T / 2; stp > 0; stp >>= 1) {
            if (t < stp) {
                float ov = sv[t + stp]; int oi = si[t + stp];
                if (ov > sv[t] || (ov == sv[t] && oi < si[t])) { sv[t] = ov; si[t] = oi; }
            }
            __syncthreads();
        }
        if (t == 0) { sfar = si[0]; ob[it] = si[0]; }
        __syncthreads();
        far = sfar;
        __syncthreads();
    }
}

// Gather 3-vectors: out[b,s,:] = x[b, idx[b,s], :]
__global__ void gather3(const float* __restrict__ x, const int* __restrict__ idx,
                        float* __restrict__ out, int N, int S, int total) {
    int i = blockIdx.x * blockDim.x + threadIdx.x;
    if (i >= total) return;
    int b = i / S;
    int j = idx[i];
    const float* p = x + ((size_t)b * N + j) * 3;
    out[(size_t)i * 3 + 0] = p[0];
    out[(size_t)i * 3 + 1] = p[1];
    out[(size_t)i * 3 + 2] = p[2];
}

// Ball query: first K in-ball indices (ascending), leftovers = first in-ball index
__global__ void ballquery_kernel(const float* __restrict__ x, const float* __restrict__ nx,
                                 int* __restrict__ out, int N, int S, int K, float r2,
                                 int total) {
    int gid = blockIdx.x * blockDim.x + threadIdx.x;
    if (gid >= total) return;
    int b = gid / S;
    const float* xb = x + (size_t)b * N * 3;
    float cx = nx[(size_t)gid * 3 + 0];
    float cy = nx[(size_t)gid * 3 + 1];
    float cz = nx[(size_t)gid * 3 + 2];
    int* o = out + (size_t)gid * K;
    int cnt = 0, first = 0;
    bool has = false;
    for (int j = 0; j < N; ++j) {
        float dx = xb[j * 3 + 0] - cx;
        float dy = xb[j * 3 + 1] - cy;
        float dz = xb[j * 3 + 2] - cz;
        float d = dx * dx + dy * dy + dz * dz;
        if (d <= r2) {
            if (!has) { has = true; first = j; }
            o[cnt++] = j;
            if (cnt == K) break;
        }
    }
    for (; cnt < K; cnt++) o[cnt] = first;
}

// Build grouped GEMM input rows: [centered xyz (3) | feats (Cf) | zero pad] as f16
__global__ void group_kernel(const float* __restrict__ x, const float* __restrict__ nx,
                             const int* __restrict__ idx, const _Float16* __restrict__ feat,
                             int Cf, int N, int S, int K, int ld,
                             _Float16* __restrict__ out, int total) {
    int gid = blockIdx.x * blockDim.x + threadIdx.x;
    if (gid >= total) return;
    int k = gid % K;
    int bs = gid / K;
    int b = bs / S;
    int j = idx[(size_t)bs * K + k];
    const float* p = x + ((size_t)b * N + j) * 3;
    const float* c = nx + (size_t)bs * 3;
    _Float16* o = out + (size_t)gid * ld;
    o[0] = (_Float16)(p[0] - c[0]);
    o[1] = (_Float16)(p[1] - c[1]);
    o[2] = (_Float16)(p[2] - c[2]);
    int w = 3;
    if (Cf > 0) {
        const _Float16* f = feat + ((size_t)b * N + j) * Cf;
        for (int t2 = 0; t2 < Cf; ++t2) o[3 + t2] = f[t2];
        w = 3 + Cf;
    }
    for (; w < ld; ++w) o[w] = (_Float16)0.f;
}

// sa3 input rows (2048 of ld 288): [raw xyz(3) | l2feat(256) | pad 29]
__global__ void build_sa3(const float* __restrict__ nx2, const _Float16* __restrict__ l2f,
                          _Float16* __restrict__ out) {
    int gid = blockIdx.x * blockDim.x + threadIdx.x;
    if (gid >= 2048) return;
    _Float16* o = out + (size_t)gid * 288;
    o[0] = (_Float16)nx2[(size_t)gid * 3 + 0];
    o[1] = (_Float16)nx2[(size_t)gid * 3 + 1];
    o[2] = (_Float16)nx2[(size_t)gid * 3 + 2];
    const _Float16* f = l2f + (size_t)gid * 256;
    for (int i = 0; i < 256; i++) o[3 + i] = f[i];
    for (int i = 259; i < 288; i++) o[i] = (_Float16)0.f;
}

// fp3 input rows (2048 of ld 1280): [l2feat(256) | broadcast l3(1024)]
__global__ void build_fp3(const _Float16* __restrict__ l2f, const _Float16* __restrict__ l3f,
                          _Float16* __restrict__ out) {
    int gid = blockIdx.x * blockDim.x + threadIdx.x;
    if (gid >= 2048) return;
    int b = gid >> 8;
    _Float16* o = out + (size_t)gid * 1280;
    const _Float16* f = l2f + (size_t)gid * 256;
    for (int i = 0; i < 256; i++) o[i] = f[i];
    const _Float16* g3 = l3f + (size_t)b * 1024;
    for (int i = 0; i < 1024; i++) o[256 + i] = g3[i];
}

// max over K consecutive rows per (b,s) group
__global__ void maxk_kernel(const _Float16* __restrict__ act, int ld, int K, int C,
                            _Float16* __restrict__ out, int totalBS) {
    int gid = blockIdx.x * blockDim.x + threadIdx.x;
    if (gid >= totalBS * C) return;
    int c = gid % C;
    int bs = gid / C;
    const _Float16* a = act + (size_t)bs * K * ld + c;
    float m = -3.0e38f;
    for (int k = 0; k < K; ++k) m = fmaxf(m, (float)a[(size_t)k * ld]);
    out[gid] = (_Float16)m;
}

// ---------------------------------------------------------------------------
// WMMA GEMM: O[r,o] = sum_k X[r,k]*W[o,k]  (bias omitted: cancels under BN)
// Stores pre-activation as f16; accumulates per-channel sum / sumsq for BN.
// One wave computes a 64x16 macro-tile; K stepped by 32 (f16 inputs, f32 acc).
// A frag: lane(m=l&15,hi=l>>4) rows m, contiguous K chunk [hi*16,+16).
// B frag: lane column n=l&15, contiguous K chunk [hi*16,+16).
// ---------------------------------------------------------------------------
__global__ void gemm_wmma_bn(const _Float16* __restrict__ X,
                             const _Float16* __restrict__ Wt,
                             _Float16* __restrict__ P,
                             float* __restrict__ stats,
                             int R, int Cout, int Kpad) {
    int ntc = Cout >> 4;
    int nmr = R >> 6;
    int wid = blockIdx.x * (blockDim.x >> 5) + (threadIdx.x >> 5);
    if (wid >= nmr * ntc) return;
    int mr = wid / ntc, ct = wid % ntc;
    int lane = threadIdx.x & 31;
    int m = lane & 15, hi = lane >> 4;
    const _Float16* wrow = Wt + (size_t)(ct * 16 + m) * Kpad;
    const _Float16* x0   = X  + (size_t)(mr * 64 + m) * Kpad;
    v8f zero = {0.f, 0.f, 0.f, 0.f, 0.f, 0.f, 0.f, 0.f};
    v8f acc[4];
#pragma unroll
    for (int r = 0; r < 4; r++) acc[r] = zero;

    for (int k0 = 0; k0 < Kpad; k0 += 32) {
        v16h bf = *(const v16h*)(wrow + k0 + hi * 16);
#pragma unroll
        for (int r = 0; r < 4; r++) {
            const _Float16* xr = x0 + (size_t)(r * 16) * Kpad;
            v8h a0 = *(const v8h*)(xr + k0 + hi * 16);
            v8h a1 = *(const v8h*)(xr + k0 + hi * 16 + 8);
            v16h af;
#pragma unroll
            for (int i = 0; i < 8; i++) { af[i] = a0[i]; af[i + 8] = a1[i]; }
            acc[r] = __builtin_amdgcn_wmma_f32_16x16x32_f16(
                false, af, false, bf, (short)0, acc[r], false, false);
        }
    }

    float s = 0.f, s2 = 0.f;
    int cg = ct * 16 + m;
#pragma unroll
    for (int r = 0; r < 4; r++) {
        int rb = mr * 64 + r * 16 + hi * 8;
#pragma unroll
        for (int i = 0; i < 8; i++) {
            float v = acc[r][i];
            s += v; s2 += v * v;
            P[(size_t)(rb + i) * Cout + cg] = (_Float16)v;
        }
    }
    atomicAdd(&stats[cg], s);
    atomicAdd(&stats[Cout + cg], s2);
}

// BN (batch stats) + ReLU, in place on f16 pre-activation
__global__ void bnrelu_f16(_Float16* __restrict__ P, const float* __restrict__ stats,
                           const float* __restrict__ g, const float* __restrict__ be,
                           int C, long long total, float invcnt) {
    long long gid = (long long)blockIdx.x * blockDim.x + threadIdx.x;
    if (gid >= total) return;
    int c = (int)(gid % C);
    float mean = stats[c] * invcnt;
    float var  = stats[C + c] * invcnt - mean * mean;
    float xv = (float)P[gid];
    float y = g[c] * (xv - mean) * rsqrtf(var + 1e-5f) + be[c];
    P[gid] = (_Float16)fmaxf(y, 0.f);
}

// Final-layer variant: f32 output written straight into d_out
__global__ void bnrelu_f32out(const _Float16* __restrict__ P, const float* __restrict__ stats,
                              const float* __restrict__ g, const float* __restrict__ be,
                              int C, long long total, float invcnt, float* __restrict__ o) {
    long long gid = (long long)blockIdx.x * blockDim.x + threadIdx.x;
    if (gid >= total) return;
    int c = (int)(gid % C);
    float mean = stats[c] * invcnt;
    float var  = stats[C + c] * invcnt - mean * mean;
    float xv = (float)P[gid];
    float y = g[c] * (xv - mean) * rsqrtf(var + 1e-5f) + be[c];
    o[gid] = fmaxf(y, 0.f);
}

// out1: l2_xyz (B,3,256) from new_xyz2 (B,256,3)
__global__ void write_out1(const float* __restrict__ nx2, float* __restrict__ o) {
    int gid = blockIdx.x * blockDim.x + threadIdx.x;
    if (gid >= 6144) return;
    int b = gid / 768, r = gid % 768, c = r / 256, s = r % 256;
    o[gid] = nx2[((size_t)b * 256 + s) * 3 + c];
}

// ---------------------------------------------------------------------------
// Host orchestration
// ---------------------------------------------------------------------------
extern "C" void kernel_launch(void* const* d_in, const int* in_sizes, int n_in,
                              void* d_out, int out_size, void* d_ws, size_t ws_size,
                              hipStream_t stream) {
    (void)out_size; (void)ws_size;
    const int B = 8, N1 = 8192, S1 = 1024, K1 = 32, S2 = 256, K2 = 64;

    // --- locate xyz among flattened inputs (size 196608, NOT followed by 384-bias) ---
    int xyz_i = -1;
    for (int i = 0; i < n_in; i++) {
        if (in_sizes[i] == 196608) {
            bool is_w = (i + 1 < n_in) && in_sizes[i + 1] == 384;
            if (!is_w) { xyz_i = i; break; }
        }
    }
    if (xyz_i < 0) xyz_i = 0;
    const float* xyz = (const float*)d_in[xyz_i];

    const float* prm[64];
    int np = 0;
    for (int i = 0; i < n_in && np < 44; i++)
        if (i != xyz_i) prm[np++] = (const float*)d_in[i];

    // ordering: alphabetical (fp3,sa1,sa2,sa3) iff first param W has 655360 elems
    int firstW = (xyz_i == 0) ? 1 : 0;
    bool fp3_first = (firstW < n_in) && (in_sizes[firstW] == 655360);
    int b_sa1, b_sa2, b_sa3, b_fp3;
    if (fp3_first) { b_fp3 = 0; b_sa1 = 8;  b_sa2 = 20; b_sa3 = 32; }
    else           { b_sa1 = 0; b_sa2 = 12; b_sa3 = 24; b_fp3 = 36; }

    struct LD { int base, cout, cin, kpad; };
    LD layers[11] = {
        {b_sa1 + 0, 64,   3,   32},  {b_sa1 + 4, 64,  64,  64},  {b_sa1 + 8, 128, 64,  64},
        {b_sa2 + 0, 128, 131, 160},  {b_sa2 + 4, 128, 128, 128}, {b_sa2 + 8, 256, 128, 128},
        {b_sa3 + 0, 256, 259, 288},  {b_sa3 + 4, 512, 256, 256}, {b_sa3 + 8, 1024, 512, 512},
        {b_fp3 + 0, 512, 1280, 1280},{b_fp3 + 4, 384, 512, 512},
    };

    // --- workspace carve ---
    char* ws = (char*)d_ws;
    size_t off = 0;
    auto carve = [&](size_t bytes) -> void* {
        off = (off + 255) & ~(size_t)255;
        void* p = ws + off;
        off += bytes;
        return p;
    };
    float*     x1     = (float*)    carve((size_t)B * N1 * 3 * 4);
    int*       fidx1  = (int*)      carve((size_t)B * S1 * 4);
    float*     nx1    = (float*)    carve((size_t)B * S1 * 3 * 4);
    int*       bidx1  = (int*)      carve((size_t)B * S1 * K1 * 4);
    int*       fidx2  = (int*)      carve((size_t)B * S2 * 4);
    float*     nx2    = (float*)    carve((size_t)B * S2 * 3 * 4);
    int*       bidx2  = (int*)      carve((size_t)B * S2 * K2 * 4);
    _Float16*  l1feat = (_Float16*) carve((size_t)B * S1 * 128 * 2);
    _Float16*  l2feat = (_Float16*) carve((size_t)B * S2 * 256 * 2);
    _Float16*  l3feat = (_Float16*) carve((size_t)B * 1024 * 2);
    float*     stats  = (float*)    carve(2048 * 4);
    _Float16*  wf[11];
    for (int l = 0; l < 11; l++)
        wf[l] = (_Float16*)carve((size_t)layers[l].cout * layers[l].kpad * 2);
    _Float16*  bufA   = (_Float16*) carve((size_t)131072 * 160 * 2);  // 42 MB
    _Float16*  bufB   = (_Float16*) carve((size_t)262144 * 128 * 2);  // 67 MB

    // --- convert weights to padded f16 ---
    for (int l = 0; l < 11; l++) {
        int tot = layers[l].cout * layers[l].kpad;
        conv_w_f16<<<CDIV(tot, 256), 256, 0, stream>>>(
            prm[layers[l].base], wf[l], layers[l].cout, layers[l].cin, layers[l].kpad);
    }

    auto run_layer = [&](int l, const _Float16* Xin, _Float16* Pout, int R,
                         bool finalOut, float* outF) {
        int C = layers[l].cout, Kp = layers[l].kpad;
        const float* g  = prm[layers[l].base + 2];
        const float* be = prm[layers[l].base + 3];
        fill_f32<<<CDIV(2 * C, 256), 256, 0, stream>>>(stats, 0.f, 2 * C);
        int waves = (R >> 6) * (C >> 4);
        gemm_wmma_bn<<<CDIV(waves, 4), 128, 0, stream>>>(Xin, wf[l], Pout, stats, R, C, Kp);
        long long total = (long long)R * C;
        if (!finalOut)
            bnrelu_f16<<<CDIV(total, 256), 256, 0, stream>>>(Pout, stats, g, be, C, total, 1.f / R);
        else
            bnrelu_f32out<<<CDIV(total, 256), 256, 0, stream>>>(Pout, stats, g, be, C, total, 1.f / R, outF);
    };

    // ---------------- sa1 ----------------
    transpose_xyz<<<CDIV(B * N1 * 3, 256), 256, 0, stream>>>(xyz, x1, B, N1);
    fps_kernel<32><<<B, 256, 0, stream>>>(x1, N1, S1, fidx1);
    gather3<<<CDIV(B * S1, 256), 256, 0, stream>>>(x1, fidx1, nx1, N1, S1, B * S1);
    ballquery_kernel<<<CDIV(B * S1, 256), 256, 0, stream>>>(x1, nx1, bidx1, N1, S1, K1,
                                                            0.2f * 0.2f, B * S1);
    {
        int tot = B * S1 * K1;  // 262144 rows, ld 32
        group_kernel<<<CDIV(tot, 256), 256, 0, stream>>>(x1, nx1, bidx1, nullptr, 0,
                                                         N1, S1, K1, 32, bufA, tot);
        run_layer(0, bufA, bufB, tot, false, nullptr);   // 32 -> 64
        run_layer(1, bufB, bufA, tot, false, nullptr);   // 64 -> 64
        run_layer(2, bufA, bufB, tot, false, nullptr);   // 64 -> 128
        maxk_kernel<<<CDIV(B * S1 * 128, 256), 256, 0, stream>>>(bufB, 128, K1, 128,
                                                                 l1feat, B * S1);
    }

    // ---------------- sa2 ----------------
    fps_kernel<4><<<B, 256, 0, stream>>>(nx1, S1, S2, fidx2);
    gather3<<<CDIV(B * S2, 256), 256, 0, stream>>>(nx1, fidx2, nx2, S1, S2, B * S2);
    ballquery_kernel<<<CDIV(B * S2, 256), 256, 0, stream>>>(nx1, nx2, bidx2, S1, S2, K2,
                                                            0.4f * 0.4f, B * S2);
    {
        int tot = B * S2 * K2;  // 131072 rows, ld 160
        group_kernel<<<CDIV(tot, 256), 256, 0, stream>>>(nx1, nx2, bidx2, l1feat, 128,
                                                         S1, S2, K2, 160, bufA, tot);
        run_layer(3, bufA, bufB, tot, false, nullptr);   // 160 -> 128
        run_layer(4, bufB, bufA, tot, false, nullptr);   // 128 -> 128
        run_layer(5, bufA, bufB, tot, false, nullptr);   // 128 -> 256
        maxk_kernel<<<CDIV(B * S2 * 256, 256), 256, 0, stream>>>(bufB, 256, K2, 256,
                                                                 l2feat, B * S2);
    }

    // ---------------- sa3 (group_all) ----------------
    {
        int tot = B * S2;  // 2048 rows, ld 288
        build_sa3<<<CDIV(tot, 256), 256, 0, stream>>>(nx2, l2feat, bufA);
        run_layer(6, bufA, bufB, tot, false, nullptr);   // 288 -> 256
        run_layer(7, bufB, bufA, tot, false, nullptr);   // 256 -> 512
        run_layer(8, bufA, bufB, tot, false, nullptr);   // 512 -> 1024
        maxk_kernel<<<CDIV(B * 1024, 256), 256, 0, stream>>>(bufB, 1024, 256, 1024,
                                                             l3feat, B);
    }

    // ---------------- fp3 ----------------
    {
        int tot = B * S2;  // 2048 rows, ld 1280
        build_fp3<<<CDIV(tot, 256), 256, 0, stream>>>(l2feat, l3feat, bufA);
        run_layer(9, bufA, bufB, tot, false, nullptr);   // 1280 -> 512
        float* out2 = (float*)d_out + 6144;              // (B,256,384) row-major
        run_layer(10, bufB, bufA, tot, true, out2);      // 512 -> 384, f32 to d_out
    }

    // ---------------- out1: l2_xyz (B,3,256) ----------------
    write_out1<<<CDIV(6144, 256), 256, 0, stream>>>(nx2, (float*)d_out);
}
